// dvs_gesture_snn_32x32_33122787787170
// MI455X (gfx1250) — matrix operand order repaired and checked by reading, not compiled, not run on hardware
//
#include <hip/hip_runtime.h>
#include <hip/hip_bf16.h>
#include <cstddef>
#include <cstdint>

typedef __attribute__((ext_vector_type(16))) _Float16 v16h;
typedef __attribute__((ext_vector_type(8)))  float    v8f;

#define BB 25
#define TT 30
#define BT 750

// ---------------- workspace layout (bytes) ----------------
#define OFF_Z0   ((size_t)0)            //  750*2*1024 f16   = 3,072,000
#define OFF_A    ((size_t)3072000)      //  750*32*1024 f16  = 49,152,000
#define OFF_B    ((size_t)52224000)     //  750*32*1024 f16  = 49,152,000
#define OFF_P3   ((size_t)101376000)    //  750*32*256 f16   = 12,288,000
#define OFF_P4   ((size_t)113664000)    //  750*64*16 f16    = 1,536,000
#define OFF_W1P  ((size_t)115200000)    //  2*1*512 f16      = 2,048
#define OFF_W2P  ((size_t)115202048)    //  2*9*512 f16      = 18,432
#define OFF_W3P  ((size_t)115220480)
#define OFF_W4P  ((size_t)115238912)    //  4*9*512 f16      = 36,864
#define OFF_W5P  ((size_t)115275776)    //  4*2*512 f16      = 8,192
#define OFF_FC1P ((size_t)115283968)    //  4*32*512 f16     = 131,072

// ---------------------------------------------------------------------------
// Pack fp32 OIHW conv weights (or [COUT][K] FC weights with KS=1,CIN=Kdim)
// into the CDNA5 WMMA 16-bit A-matrix VGPR layout, per (mtile,ktile):
//   lanes 0-15 : M=lane,    half h -> K = (h<8)? h   : h+8
//   lanes 16-31: M=lane-16, half h -> K = (h<8)? h+8 : h+16
// ---------------------------------------------------------------------------
__global__ void pack_w_kernel(const float* __restrict__ w, _Float16* __restrict__ wp,
                              int COUT, int CIN, int KS) {
    const int Kdim = CIN * KS * KS;
    const int KT = (Kdim + 31) >> 5;
    const int total = (COUT / 16) * KT * 512;
    int i = blockIdx.x * blockDim.x + threadIdx.x;
    if (i >= total) return;
    int h    = i & 15;
    int lane = (i >> 4) & 31;
    int mk   = i >> 9;            // mt*KT + kt
    int kt   = mk % KT;
    int mt   = mk / KT;
    int M, K;
    if (lane < 16) { M = lane;      K = (h < 8) ? h     : h + 8;  }
    else           { M = lane - 16; K = (h < 8) ? h + 8 : h + 16; }
    int gk = kt * 32 + K;
    int oc = mt * 16 + M;
    float v = 0.f;
    if (gk < Kdim) v = w[(size_t)oc * Kdim + gk];
    wp[i] = (_Float16)v;
}

// ---------------------------------------------------------------------------
// 4x4 average pool on fp32 input -> f16, (BT,2,128,128) -> (BT,2,32,32)
// ---------------------------------------------------------------------------
__global__ void pool_in4_kernel(const float* __restrict__ x, _Float16* __restrict__ z,
                                int total) {
    int i = blockIdx.x * blockDim.x + threadIdx.x;
    if (i >= total) return;
    int xo   = i & 31;
    int yo   = (i >> 5) & 31;
    int rest = i >> 10;                       // img*2 + c
    const float* src = x + (size_t)rest * 128 * 128 + (size_t)(yo * 4) * 128 + xo * 4;
    float s = 0.f;
    #pragma unroll
    for (int a = 0; a < 4; ++a)
        #pragma unroll
        for (int b = 0; b < 4; ++b) s += src[a * 128 + b];
    z[i] = (_Float16)(s * 0.0625f);
}

// generic KKxKK avg pool on f16 maps (all extents compile-time)
template<int KK, int HIN, int WIN>
__global__ void pool_f16_kernel(const _Float16* __restrict__ in, _Float16* __restrict__ out,
                                int total) {
    constexpr int WOUT = WIN / KK, HOUT = HIN / KK;
    int i = blockIdx.x * blockDim.x + threadIdx.x;
    if (i >= total) return;
    int xo = i % WOUT;
    int yo = (i / WOUT) % HOUT;
    int rest = i / (WOUT * HOUT);             // img*C + c
    const _Float16* src = in + (size_t)rest * HIN * WIN + (size_t)(yo * KK) * WIN + xo * KK;
    float s = 0.f;
    #pragma unroll
    for (int a = 0; a < KK; ++a)
        #pragma unroll
        for (int b = 0; b < KK; ++b) s += (float)src[a * WIN + b];
    out[i] = (_Float16)(s * (1.f / (KK * KK)));
}

// ---------------------------------------------------------------------------
// Implicit-im2col conv as WMMA GEMM.
//   block = NWAVES waves; wave w owns M-tile w (16 output channels); all waves
//   share one 32Kx16N B tile, cooperatively gathered into LDS in the WMMA
//   B-fragment layout (lane L<16 holds K=0..7,16..23 of N=L; L>=16 holds K=8..15,24..31).
//   Packed A (weights) for each wave's M-tile is DMA'd global->LDS with
//   gfx1250 async-to-LDS copies issued up front; first B gather overlaps the
//   in-flight DMA; s_wait_asynccnt 0 before the first A fragment read.
//   grid = (HW/16, 1, BT)
// ---------------------------------------------------------------------------
template<int CIN, int KS, int PAD, int H, int W, int NWAVES>
__global__ void __launch_bounds__(32 * NWAVES)
conv_wmma_kernel(const _Float16* __restrict__ in, const _Float16* __restrict__ wp,
                 const float* __restrict__ scale, const float* __restrict__ offs,
                 _Float16* __restrict__ out) {
    constexpr int HW   = H * W;
    constexpr int Kdim = CIN * KS * KS;
    constexpr int KT   = (Kdim + 31) >> 5;
    constexpr int COUT = 16 * NWAVES;
    constexpr int NTH  = 32 * NWAVES;

    __shared__ __align__(32) _Float16 lB[512];
    __shared__ __align__(16) _Float16 lAw[NWAVES * KT * 512];

    const int img = blockIdx.z, nt = blockIdx.x;
    const int tid  = threadIdx.x;
    const int lane = tid & 31;
    const int mt   = tid >> 5;                 // wave id == M-tile
    const int p0   = nt * 16;
    const _Float16* inImg = in + (size_t)img * CIN * HW;

    // ---- issue async DMA of this wave's packed weights into LDS ----
    {
        const char* gsrc = (const char*)(wp + (size_t)mt * KT * 512);
        char*       ldst = (char*)&lAw[(size_t)mt * KT * 512];
        #pragma unroll
        for (int c = 0; c < KT * 2; ++c) {          // KT*64 16B chunks / 32 lanes
            int idx = c * 32 + lane;
            uint64_t ga = (uint64_t)(uintptr_t)(gsrc + (size_t)idx * 16);
            uint32_t la = (uint32_t)(uintptr_t)(ldst + (size_t)idx * 16);
            asm volatile("global_load_async_to_lds_b128 %0, %1, off"
                         :: "v"(la), "v"(ga) : "memory");
        }
    }

    v8f acc = {};
    for (int kt = 0; kt < KT; ++kt) {
        // --- cooperative im2col gather of the shared B tile ---
        #pragma unroll
        for (int e0 = 0; e0 < 512 / NTH; ++e0) {
            int e = e0 * NTH + tid;
            int K = e >> 4, n = e & 15;
            int gk = kt * 32 + K;
            _Float16 val = (_Float16)0.f;
            if (gk < Kdim) {
                int ic  = gk / (KS * KS);
                int tap = gk % (KS * KS);
                int y = (p0 + n) / W + tap / KS - PAD;
                int x = (p0 + n) % W + tap % KS - PAD;
                if ((unsigned)y < (unsigned)H && (unsigned)x < (unsigned)W)
                    val = inImg[(size_t)ic * HW + (size_t)y * W + x];
            }
            lB[(n + (((K >> 3) & 1) ? 16 : 0)) * 16 + (K & 7) + ((K >> 4) << 3)] = val;
        }
        if (kt == 0)   // weight DMA must have landed before first A read
            asm volatile("s_wait_asynccnt 0" ::: "memory");
        __syncthreads();
        v16h a = *(const v16h*)(&lAw[(size_t)(mt * KT + kt) * 512 + lane * 16]);
        v16h b = *(const v16h*)(&lB[lane * 16]);
        acc = __builtin_amdgcn_wmma_f32_16x16x32_f16(
            false, a, false, b, (short)0, acc, false, false);
        __syncthreads();
    }

    // epilogue: folded BN, store f16
    const int N = lane & 15;
    const int Mbase = (lane >= 16) ? 8 : 0;
    _Float16* outImg = out + (size_t)img * COUT * HW;
    #pragma unroll
    for (int r = 0; r < 8; ++r) {
        int oc = mt * 16 + Mbase + r;
        float v = acc[r] * scale[oc] + offs[oc];
        outImg[(size_t)oc * HW + (p0 + N)] = (_Float16)v;
    }
}

// ---------------------------------------------------------------------------
// FC GEMM via WMMA with register-direct B:
// B-fragment lane L holds K 8h..8h+7 and 16+8h..16+8h+7 (h=L>>4) of column N=L&15, loaded as
// two 16B global loads from in[row][kt*32 + (L>>4)*8] and 16 halves further on. No LDS at all.
//   grid = (ceil(Nrows/16), COUT/16), block = 32
// ---------------------------------------------------------------------------
__global__ void __launch_bounds__(32)
fc_wmma_kernel(const _Float16* __restrict__ in, const _Float16* __restrict__ wp,
               _Float16* __restrict__ out, int Nrows, int Kdim, int COUT) {
    const int nt = blockIdx.x, mt = blockIdx.y;
    const int lane = threadIdx.x;
    const int KT = Kdim >> 5;
    const int n0 = nt * 16;
    const int N  = lane & 15;
    const int kh = lane >> 4;                  // K-half: 0 -> K 0..7,16..23; 1 -> K 8..15,24..31
    const int row = n0 + N;
    const bool vr = row < Nrows;
    const _Float16* rp = in + (size_t)(vr ? row : 0) * Kdim + kh * 8;

    v8f acc = {};
    for (int kt = 0; kt < KT; ++kt) {
        if (kt + 1 < KT)
            __builtin_prefetch(wp + ((size_t)(mt * KT + kt + 1) * 32 + lane) * 16, 0, 0);
        v16h a = *(const v16h*)(wp + ((size_t)(mt * KT + kt) * 32 + lane) * 16);
        v16h b = {};
        if (vr) {
            typedef __attribute__((ext_vector_type(8))) _Float16 v8h_t;
            v8h_t b0 = *(const v8h_t*)(rp + (size_t)kt * 32);
            v8h_t b1 = *(const v8h_t*)(rp + (size_t)kt * 32 + 16);
            b = __builtin_shufflevector(b0, b1, 0,1,2,3,4,5,6,7,8,9,10,11,12,13,14,15);
        }
        acc = __builtin_amdgcn_wmma_f32_16x16x32_f16(
            false, a, false, b, (short)0, acc, false, false);
    }
    const int Mbase = (lane >= 16) ? 8 : 0;
    #pragma unroll
    for (int r = 0; r < 8; ++r) {
        int oc = mt * 16 + Mbase + r;
        if (vr) out[(size_t)row * COUT + oc] = (_Float16)acc[r];
    }
}

// ---------------------------------------------------------------------------
// IAF time scan: v += c_t ; s = max(floor(v),0) ; v = max(v - s, -1)
// grid = (ceil(F/256), B); one thread per (batch, feature), T=30 serial steps
// ---------------------------------------------------------------------------
__global__ void iaf_kernel(const _Float16* __restrict__ c, _Float16* __restrict__ s, int F) {
    int f = blockIdx.x * blockDim.x + threadIdx.x;
    if (f >= F) return;
    int b = blockIdx.y;
    size_t base = (size_t)b * TT * F + f;
    float v = 0.f;
    for (int t = 0; t < TT; ++t) {
        v += (float)c[base + (size_t)t * F];
        float sp = fmaxf(floorf(v), 0.f);
        s[base + (size_t)t * F] = (_Float16)sp;
        v = fmaxf(v - sp, -1.f);
    }
}

// ---------------------------------------------------------------------------
// Final FC 64 -> 11 (too small for WMMA), fp32 output
// ---------------------------------------------------------------------------
__global__ void fc2_kernel(const _Float16* __restrict__ s, const float* __restrict__ w,
                           float* __restrict__ out, int total) {
    int i = blockIdx.x * blockDim.x + threadIdx.x;
    if (i >= total) return;
    int o = i % 11, n = i / 11;
    float sum = 0.f;
    for (int k = 0; k < 64; ++k)
        sum += (float)s[(size_t)n * 64 + k] * w[(size_t)o * 64 + k];
    out[i] = sum;
}

// ---------------------------------------------------------------------------
extern "C" void kernel_launch(void* const* d_in, const int* in_sizes, int n_in,
                              void* d_out, int out_size, void* d_ws, size_t ws_size,
                              hipStream_t stream) {
    const float* x   = (const float*)d_in[0];
    const float* w1  = (const float*)d_in[1];
    const float* w2  = (const float*)d_in[2];
    const float* w3  = (const float*)d_in[3];
    const float* w4  = (const float*)d_in[4];
    const float* w5  = (const float*)d_in[5];
    const float* s1  = (const float*)d_in[6];
    const float* s2  = (const float*)d_in[7];
    const float* s3  = (const float*)d_in[8];
    const float* s4  = (const float*)d_in[9];
    const float* s5  = (const float*)d_in[10];
    const float* o1  = (const float*)d_in[11];
    const float* o2  = (const float*)d_in[12];
    const float* o3  = (const float*)d_in[13];
    const float* o4  = (const float*)d_in[14];
    const float* o5  = (const float*)d_in[15];
    const float* fc1 = (const float*)d_in[16];
    const float* fc2 = (const float*)d_in[17];
    float* out = (float*)d_out;

    char* ws = (char*)d_ws;
    _Float16* z0   = (_Float16*)(ws + OFF_Z0);
    _Float16* bufA = (_Float16*)(ws + OFF_A);
    _Float16* bufB = (_Float16*)(ws + OFF_B);
    _Float16* p3   = (_Float16*)(ws + OFF_P3);
    _Float16* p4   = (_Float16*)(ws + OFF_P4);
    _Float16* w1p  = (_Float16*)(ws + OFF_W1P);
    _Float16* w2p  = (_Float16*)(ws + OFF_W2P);
    _Float16* w3p  = (_Float16*)(ws + OFF_W3P);
    _Float16* w4p  = (_Float16*)(ws + OFF_W4P);
    _Float16* w5p  = (_Float16*)(ws + OFF_W5P);
    _Float16* fc1p = (_Float16*)(ws + OFF_FC1P);

    // ---- pack weights into WMMA A-fragment layout ----
    {
        int t;
        t = (32 / 16) * 1 * 512;   // w1: CIN=2, KS=3 -> K=18 -> KT=1
        pack_w_kernel<<<(t + 255) / 256, 256, 0, stream>>>(w1, w1p, 32, 2, 3);
        t = (32 / 16) * 9 * 512;   // w2/w3: K=288 -> KT=9
        pack_w_kernel<<<(t + 255) / 256, 256, 0, stream>>>(w2, w2p, 32, 32, 3);
        pack_w_kernel<<<(t + 255) / 256, 256, 0, stream>>>(w3, w3p, 32, 32, 3);
        t = (64 / 16) * 9 * 512;   // w4
        pack_w_kernel<<<(t + 255) / 256, 256, 0, stream>>>(w4, w4p, 64, 32, 3);
        t = (64 / 16) * 2 * 512;   // w5: 1x1, K=64 -> KT=2
        pack_w_kernel<<<(t + 255) / 256, 256, 0, stream>>>(w5, w5p, 64, 64, 1);
        t = (64 / 16) * 32 * 512;  // fc1: K=1024 -> KT=32
        pack_w_kernel<<<(t + 255) / 256, 256, 0, stream>>>(fc1, fc1p, 64, 1024, 1);
    }

    // ---- stage 0: 4x4 pool of input ----
    {
        int t = BT * 2 * 32 * 32;
        pool_in4_kernel<<<(t + 255) / 256, 256, 0, stream>>>(x, z0, t);
    }

    // ---- conv1 + IAF ----
    conv_wmma_kernel<2, 3, 1, 32, 32, 2><<<dim3(64, 1, BT), 64, 0, stream>>>(
        z0, w1p, s1, o1, bufA);
    {
        int F = 32 * 1024;
        iaf_kernel<<<dim3((F + 255) / 256, BB), 256, 0, stream>>>(bufA, bufB, F);
    }
    // ---- conv2 + IAF ----
    conv_wmma_kernel<32, 3, 1, 32, 32, 2><<<dim3(64, 1, BT), 64, 0, stream>>>(
        bufB, w2p, s2, o2, bufA);
    {
        int F = 32 * 1024;
        iaf_kernel<<<dim3((F + 255) / 256, BB), 256, 0, stream>>>(bufA, bufB, F);
    }
    // ---- conv3 + IAF ----
    conv_wmma_kernel<32, 3, 1, 32, 32, 2><<<dim3(64, 1, BT), 64, 0, stream>>>(
        bufB, w3p, s3, o3, bufA);
    {
        int F = 32 * 1024;
        iaf_kernel<<<dim3((F + 255) / 256, BB), 256, 0, stream>>>(bufA, bufB, F);
    }
    // ---- 2x2 pool ----
    {
        int t = BT * 32 * 16 * 16;
        pool_f16_kernel<2, 32, 32><<<(t + 255) / 256, 256, 0, stream>>>(bufB, p3, t);
    }
    // ---- conv4 + IAF ----
    conv_wmma_kernel<32, 3, 1, 16, 16, 4><<<dim3(16, 1, BT), 128, 0, stream>>>(
        p3, w4p, s4, o4, bufA);
    {
        int F = 64 * 256;
        iaf_kernel<<<dim3((F + 255) / 256, BB), 256, 0, stream>>>(bufA, bufB, F);
    }
    // ---- 4x4 pool ----
    {
        int t = BT * 64 * 4 * 4;
        pool_f16_kernel<4, 16, 16><<<(t + 255) / 256, 256, 0, stream>>>(bufB, p4, t);
    }
    // ---- conv5 (1x1) + IAF ----
    conv_wmma_kernel<64, 1, 0, 4, 4, 4><<<dim3(1, 1, BT), 128, 0, stream>>>(
        p4, w5p, s5, o5, bufA);
    {
        int F = 64 * 16;
        iaf_kernel<<<dim3((F + 255) / 256, BB), 256, 0, stream>>>(bufA, bufB, F);
    }
    // ---- fc1 (1024 -> 64) + IAF ----
    fc_wmma_kernel<<<dim3(47, 4, 1), 32, 0, stream>>>(bufB, fc1p, bufA, BT, 1024, 64);
    {
        int F = 64;
        iaf_kernel<<<dim3((F + 255) / 256, BB), 256, 0, stream>>>(bufA, bufB, F);
    }
    // ---- fc2 (64 -> 11) -> fp32 output ----
    {
        int t = BT * 11;
        fc2_kernel<<<(t + 255) / 256, 256, 0, stream>>>(bufB, fc2, out, t);
    }
}
